// Cross_Attention_53120155517117
// MI455X (gfx1250) — compile-verified, matrix operand order checked by reading, not run on hardware
//
#include <hip/hip_runtime.h>
#include <hip/hip_bf16.h>

#define M_Q 65536
#define N_K 65536
#define KNN 16
#define CH  64

typedef _Float16 v16h __attribute__((ext_vector_type(16)));
typedef _Float16 v8h  __attribute__((ext_vector_type(8)));
typedef float    v8f  __attribute__((ext_vector_type(8)));

__device__ __forceinline__ v8f wmma16(v16h a, v16h b, v8f c) {
  return __builtin_amdgcn_wmma_f32_16x16x32_f16(
      /*neg_a=*/false, a, /*neg_b=*/false, b,
      /*c_mod=*/(short)0, c, /*reuse_a=*/false, /*reuse_b=*/false);
}

// Pack 16 f32 (two contiguous 8-element runs) into the 16-bit A-fragment
// lane layout: elements j=0..7 -> c0+j, j=8..15 -> c0+16+(j-8).
__device__ __forceinline__ v16h pack16(float4 a, float4 b, float4 c, float4 d) {
  v16h r;
  r[0]=(_Float16)a.x; r[1]=(_Float16)a.y; r[2]=(_Float16)a.z; r[3]=(_Float16)a.w;
  r[4]=(_Float16)b.x; r[5]=(_Float16)b.y; r[6]=(_Float16)b.z; r[7]=(_Float16)b.w;
  r[8]=(_Float16)c.x; r[9]=(_Float16)c.y; r[10]=(_Float16)c.z; r[11]=(_Float16)c.w;
  r[12]=(_Float16)d.x; r[13]=(_Float16)d.y; r[14]=(_Float16)d.z; r[15]=(_Float16)d.w;
  return r;
}

// B fragment from LDS-resident W^T (f16, [n][k] layout). Column n, K-run base kk0.
__device__ __forceinline__ v16h load_bfrag(const _Float16* sWT, int n, int kk0) {
  v8h lo = *(const v8h*)(sWT + n*64 + kk0);
  v8h hi = *(const v8h*)(sWT + n*64 + kk0 + 16);
  v16h b;
#pragma unroll
  for (int i = 0; i < 8; ++i) { b[i] = lo[i]; b[i+8] = hi[i]; }
  return b;
}

__device__ __forceinline__ float4 add4(float4 a, float4 b) {
  return make_float4(a.x+b.x, a.y+b.y, a.z+b.z, a.w+b.w);
}

// -------------------------------------------------------------------------
// Y[rows,64] = (X1 (+ X2)) @ W[64,64] + bias.   One wave = 16 rows.
// -------------------------------------------------------------------------
__global__ __launch_bounds__(256) void gemm64_kernel(
    const float* __restrict__ X1, const float* __restrict__ X2,
    const float* __restrict__ W,  const float* __restrict__ bias,
    float* __restrict__ Y)
{
  __shared__ _Float16 sWT[64*64];
  __shared__ float    sB[64];
  const int tid = threadIdx.x;

  for (int e = tid; e < 64*64; e += 256) {
    int kk = e >> 6, n = e & 63;
    sWT[n*64 + kk] = (_Float16)W[e];
  }
  if (tid < 64) sB[tid] = bias[tid];
  __syncthreads();

  const int wave = tid >> 5, lane = tid & 31;
  const int r = lane & 15, half = lane >> 4;
  const long rowBase = (long)blockIdx.x * 128 + wave * 16;
  const float* x1 = X1 + (rowBase + r) * 64;
  const float* x2 = X2 ? (X2 + (rowBase + r) * 64) : nullptr;

  v16h A[2];
#pragma unroll
  for (int ks = 0; ks < 2; ++ks) {
    const int c0 = ks*32 + half*8;
    float4 a0 = *(const float4*)(x1 + c0);
    float4 a1 = *(const float4*)(x1 + c0 + 4);
    float4 a2 = *(const float4*)(x1 + c0 + 16);
    float4 a3 = *(const float4*)(x1 + c0 + 20);
    if (x2) {
      a0 = add4(a0, *(const float4*)(x2 + c0));
      a1 = add4(a1, *(const float4*)(x2 + c0 + 4));
      a2 = add4(a2, *(const float4*)(x2 + c0 + 16));
      a3 = add4(a3, *(const float4*)(x2 + c0 + 20));
    }
    A[ks] = pack16(a0, a1, a2, a3);
  }

#pragma unroll
  for (int nt = 0; nt < 4; ++nt) {
    const int n = nt*16 + r;
    v8f acc;
    const float bb = sB[n];
#pragma unroll
    for (int i = 0; i < 8; ++i) acc[i] = bb;
    acc = wmma16(A[0], load_bfrag(sWT, n,      half*8), acc);
    acc = wmma16(A[1], load_bfrag(sWT, n, 32 + half*8), acc);
#pragma unroll
    for (int i = 0; i < 8; ++i)
      Y[(rowBase + i + 8*half)*64 + n] = acc[i];
  }
}

// -------------------------------------------------------------------------
// Fused per-edge pipeline. One wave = 4 queries, one at a time (16 neighbors x 64 ch tile each):
//   E = qp[m] - kp[idx[m,:]]        (gather, fp32 diff -> f16 A-frags)
//   H = relu(E @ Wg1 + bg1)         (WMMA, C-layout -> LDS -> A-frags)
//   e = H @ Wg2 + bg2; masked       (WMMA)
//   attn = softmax_k(e)             (8 regs + shfl_xor(16) per column)
//   V = value[m] @ Wv + bv          (WMMA)
//   res[m,c] = sum_k attn*V         (VALU + shfl_xor(16))
// -------------------------------------------------------------------------
__global__ __launch_bounds__(256) void edge_kernel(
    const float* __restrict__ qp, const float* __restrict__ kp,
    const float* __restrict__ value, const unsigned char* __restrict__ mask,
    const int* __restrict__ nidx,
    const float* __restrict__ Wg1, const float* __restrict__ bg1,
    const float* __restrict__ Wg2, const float* __restrict__ bg2,
    const float* __restrict__ Wv,  const float* __restrict__ bvv,
    float* __restrict__ res)
{
  __shared__ _Float16 sW1[64*64];
  __shared__ _Float16 sW2[64*64];
  __shared__ _Float16 sWv[64*64];
  __shared__ float sB1[64], sB2[64], sBv[64];
  __shared__ float sH[8][16*64];     // per-wave C-layout -> A-layout bounce

  const int tid = threadIdx.x;
  for (int e = tid; e < 64*64; e += 256) {
    int kk = e >> 6, n = e & 63;
    sW1[n*64 + kk] = (_Float16)Wg1[e];
    sW2[n*64 + kk] = (_Float16)Wg2[e];
    sWv[n*64 + kk] = (_Float16)Wv[e];
  }
  if (tid < 64) { sB1[tid]=bg1[tid]; sB2[tid]=bg2[tid]; sBv[tid]=bvv[tid]; }
  __syncthreads();

  const int wave = tid >> 5, lane = tid & 31;
  const int r = lane & 15, half = lane >> 4;
  float* hbuf = sH[wave];

  const int QPW = 4;   // queries per wave
  const long q0 = ((long)blockIdx.x * 8 + wave) * QPW;

  for (int qi = 0; qi < QPW; ++qi) {
    const long m  = q0 + qi;
    const int nb  = nidx[m*KNN + r];                 // neighbor for row r
    const float* krow = kp + (long)nb * 64;
    const float* qrow = qp + m * 64;

    // ---- E = qp - gathered kp, as two f16 A-fragments -------------------
    v16h Ea[2];
#pragma unroll
    for (int ks = 0; ks < 2; ++ks) {
      const int c0 = ks*32 + half*8;
      float4 q0v = *(const float4*)(qrow + c0);
      float4 q1v = *(const float4*)(qrow + c0 + 4);
      float4 q2v = *(const float4*)(qrow + c0 + 16);
      float4 q3v = *(const float4*)(qrow + c0 + 20);
      float4 k0v = *(const float4*)(krow + c0);
      float4 k1v = *(const float4*)(krow + c0 + 4);
      float4 k2v = *(const float4*)(krow + c0 + 16);
      float4 k3v = *(const float4*)(krow + c0 + 20);
      Ea[ks] = pack16(make_float4(q0v.x-k0v.x, q0v.y-k0v.y, q0v.z-k0v.z, q0v.w-k0v.w),
                      make_float4(q1v.x-k1v.x, q1v.y-k1v.y, q1v.z-k1v.z, q1v.w-k1v.w),
                      make_float4(q2v.x-k2v.x, q2v.y-k2v.y, q2v.z-k2v.z, q2v.w-k2v.w),
                      make_float4(q3v.x-k3v.x, q3v.y-k3v.y, q3v.z-k3v.z, q3v.w-k3v.w));
    }

    // ---- H = relu(E @ Wg1 + bg1) -> LDS ---------------------------------
#pragma unroll
    for (int nt = 0; nt < 4; ++nt) {
      const int n = nt*16 + r;
      v8f acc;
      const float bb = sB1[n];
#pragma unroll
      for (int i = 0; i < 8; ++i) acc[i] = bb;
      acc = wmma16(Ea[0], load_bfrag(sW1, n,      half*8), acc);
      acc = wmma16(Ea[1], load_bfrag(sW1, n, 32 + half*8), acc);
#pragma unroll
      for (int i = 0; i < 8; ++i)
        hbuf[(i + 8*half)*64 + n] = fmaxf(acc[i], 0.0f);
    }

    // ---- re-fragment H (A layout) from LDS ------------------------------
    v16h Ha[2];
#pragma unroll
    for (int ks = 0; ks < 2; ++ks) {
      const int c0 = ks*32 + half*8;
      const float* hr = hbuf + r*64;
      Ha[ks] = pack16(*(const float4*)(hr + c0),
                      *(const float4*)(hr + c0 + 4),
                      *(const float4*)(hr + c0 + 16),
                      *(const float4*)(hr + c0 + 20));
    }

    // ---- e = H @ Wg2 + bg2 ---------------------------------------------
    v8f eacc[4];
#pragma unroll
    for (int nt = 0; nt < 4; ++nt) {
      const int n = nt*16 + r;
      const float bb = sB2[n];
#pragma unroll
      for (int i = 0; i < 8; ++i) eacc[nt][i] = bb;
      eacc[nt] = wmma16(Ha[0], load_bfrag(sW2, n,      half*8), eacc[nt]);
      eacc[nt] = wmma16(Ha[1], load_bfrag(sW2, n, 32 + half*8), eacc[nt]);
    }

    // ---- mask (per neighbor, broadcast over channels) -------------------
    uint2 mm = *(const uint2*)(mask + m*KNN + half*8);
#pragma unroll
    for (int i = 0; i < 8; ++i) {
      const unsigned int w = (i < 4) ? mm.x : mm.y;
      const bool msk = ((w >> (8*(i & 3))) & 0xFFu) != 0u;
      if (msk) {
#pragma unroll
        for (int nt = 0; nt < 4; ++nt) eacc[nt][i] = -1e12f;
      }
    }

    // ---- per-channel softmax over the 16 neighbors ----------------------
#pragma unroll
    for (int nt = 0; nt < 4; ++nt) {
      float mx = eacc[nt][0];
#pragma unroll
      for (int i = 1; i < 8; ++i) mx = fmaxf(mx, eacc[nt][i]);
      mx = fmaxf(mx, __shfl_xor(mx, 16, 32));
      float s = 0.0f;
#pragma unroll
      for (int i = 0; i < 8; ++i) {
        float ex = __expf(eacc[nt][i] - mx);
        eacc[nt][i] = ex;
        s += ex;
      }
      s += __shfl_xor(s, 16, 32);
      const float inv = 1.0f / s;
#pragma unroll
      for (int i = 0; i < 8; ++i) eacc[nt][i] *= inv;   // attn weights
    }

    // ---- V = value[m] @ Wv + bv ; res = sum_k attn * V ------------------
    const float* vrow = value + (m*KNN + r) * 64;
    v16h Va[2];
#pragma unroll
    for (int ks = 0; ks < 2; ++ks) {
      const int c0 = ks*32 + half*8;
      Va[ks] = pack16(*(const float4*)(vrow + c0),
                      *(const float4*)(vrow + c0 + 4),
                      *(const float4*)(vrow + c0 + 16),
                      *(const float4*)(vrow + c0 + 20));
    }
#pragma unroll
    for (int nt = 0; nt < 4; ++nt) {
      const int n = nt*16 + r;
      v8f vacc;
      const float bb = sBv[n];
#pragma unroll
      for (int i = 0; i < 8; ++i) vacc[i] = bb;
      vacc = wmma16(Va[0], load_bfrag(sWv, n,      half*8), vacc);
      vacc = wmma16(Va[1], load_bfrag(sWv, n, 32 + half*8), vacc);
      float agg = 0.0f;
#pragma unroll
      for (int i = 0; i < 8; ++i) agg += eacc[nt][i] * vacc[i];
      agg += __shfl_xor(agg, 16, 32);
      if (half == 0) res[m*64 + n] = agg;
    }
  }
}

// -------------------------------------------------------------------------
extern "C" void kernel_launch(void* const* d_in, const int* in_sizes, int n_in,
                              void* d_out, int out_size, void* d_ws, size_t ws_size,
                              hipStream_t stream) {
  (void)in_sizes; (void)n_in; (void)out_size; (void)ws_size;
  const float* q     = (const float*)d_in[0];
  const float* k     = (const float*)d_in[1];
  const float* value = (const float*)d_in[2];
  const float* q_pos = (const float*)d_in[3];
  const float* k_pos = (const float*)d_in[4];
  const unsigned char* mask = (const unsigned char*)d_in[5];
  const int* knidx   = (const int*)d_in[6];
  // d_in[7] = k_num (compile-time 16)
  const float* Wqk = (const float*)d_in[8];   const float* bqk = (const float*)d_in[9];
  const float* Wv  = (const float*)d_in[10];  const float* bvv = (const float*)d_in[11];
  const float* Wg1 = (const float*)d_in[12];  const float* bg1 = (const float*)d_in[13];
  const float* Wg2 = (const float*)d_in[14];  const float* bg2 = (const float*)d_in[15];
  const float* Wt  = (const float*)d_in[16];  const float* bt  = (const float*)d_in[17];
  float* out = (float*)d_out;

  float* qp  = (float*)d_ws;                       // [M,64]
  float* kp  = qp + (size_t)M_Q * 64;              // [N,64]
  float* rs  = kp + (size_t)N_K * 64;              // [M,64]

  // qp = (q + q_pos) @ Wqk + bqk   (tied weights with kp)
  gemm64_kernel<<<M_Q/128, 256, 0, stream>>>(q, q_pos, Wqk, bqk, qp);
  // kp = (k + k_pos) @ Wqk + bqk
  gemm64_kernel<<<N_K/128, 256, 0, stream>>>(k, k_pos, Wqk, bqk, kp);
  // fused edge MLP + masked per-channel softmax + value proj + aggregate
  edge_kernel<<<M_Q/32, 256, 0, stream>>>(qp, kp, value, mask,
                                          knidx + (size_t)M_Q * KNN,  // row 1
                                          Wg1, bg1, Wg2, bg2, Wv, bvv, rs);
  // out = res @ Wt + bt
  gemm64_kernel<<<M_Q/128, 256, 0, stream>>>(rs, nullptr, Wt, bt, out);
}
